// AttentionPooling_vectorized_46059229282480
// MI455X (gfx1250) — matrix operand order repaired and checked by reading, not compiled, not run on hardware
//
#include <hip/hip_runtime.h>

typedef __bf16 bf16;
typedef __attribute__((ext_vector_type(16))) __bf16 bf16x16;
typedef __attribute__((ext_vector_type(8)))  __bf16 bf16x8;
typedef __attribute__((ext_vector_type(8)))  float  f32x8;

#define HDIM 768
#define NHEAD 4
#define DHEAD 192
#define FFN_DIM 3072
#define SPAN_L 32
#define SS 512
#define ROWS 4096   // B*N == B*S

// ---------------------------------------------------------------- utilities
__global__ void cvt_f32_bf16(const float* __restrict__ s, bf16* __restrict__ d, int n) {
    int i = blockIdx.x * 256 + threadIdx.x;
    if (i < n) d[i] = (bf16)s[i];
}

// x = token_reps + sinusoidal PE   (one block per (b,s) row)
__global__ void add_pe_kernel(const float* __restrict__ tok, float* __restrict__ xf) {
    int row = blockIdx.x;           // b*512 + s
    int s = row & (SS - 1);
    const float c = -9.210340371976184f / (float)HDIM;   // -ln(10000)/H
#pragma unroll
    for (int i = 0; i < 3; ++i) {
        int j = threadIdx.x + 256 * i;
        int i2 = j >> 1;
        float div = expf((float)(2 * i2) * c);
        float ang = (float)s * div;
        float pe = (j & 1) ? cosf(ang) : sinf(ang);
        xf[(size_t)row * HDIM + j] = tok[(size_t)row * HDIM + j] + pe;
    }
}

// q[j] = dq . Wq[j,:] + bq[j]
__global__ void qproj_kernel(const float* __restrict__ dq, const float* __restrict__ ipw,
                             const float* __restrict__ ipb, float* __restrict__ q) {
    int j = blockIdx.x * 256 + threadIdx.x;
    if (j >= HDIM) return;
    float s = ipb[j];
    const float* w = ipw + (size_t)j * HDIM;
    for (int e = 0; e < HDIM; ++e) s += dq[e] * w[e];
    q[j] = s;
}

// w_score[h,e] = sum_d q[h*192+d] * Wk[h*192+d, e]   (Wk = in_proj_w rows H..2H)
__global__ void wscore_kernel(const float* __restrict__ q, const float* __restrict__ ipw,
                              float* __restrict__ ws) {
    int t = blockIdx.x * 256 + threadIdx.x;
    if (t >= NHEAD * HDIM) return;
    int h = t / HDIM, e = t % HDIM;
    const float* wk = ipw + (size_t)HDIM * HDIM;
    float s = 0.f;
    for (int d = 0; d < DHEAD; ++d)
        s += q[h * DHEAD + d] * wk[(size_t)(h * DHEAD + d) * HDIM + e];
    ws[t] = s;
}

// --------------------------------------------------- attention pooling (per span)
__global__ __launch_bounds__(256)
void attn_pool_kernel(const float* __restrict__ xf, const int* __restrict__ span_ids,
                      const unsigned char* __restrict__ masks,
                      const float* __restrict__ wscore, bf16* __restrict__ pooled) {
    __shared__ bf16  rows[SPAN_L * HDIM];   // 48 KB
    __shared__ float sc[NHEAD * SPAN_L];
    __shared__ float at[NHEAD * SPAN_L];

    int bn = blockIdx.x, tid = threadIdx.x;
    int b = bn >> 9;                 // N == 512
    int start = span_ids[bn * 2 + 0];
    int end   = span_ids[bn * 2 + 1];
    int len   = masks[bn] ? (end - start) : 0;
    len = len < 0 ? 0 : (len > SPAN_L ? SPAN_L : len);

    // stage 32 gathered token rows into LDS (coalesced)
#pragma unroll
    for (int i = 0; i < (SPAN_L * HDIM) / 256; ++i) {
        int g = tid + 256 * i;
        int l = g / HDIM, c2 = g % HDIM;
        int r = start + l; r = r < 0 ? 0 : (r > SS - 1 ? SS - 1 : r);
        rows[g] = (bf16)xf[((size_t)(b * SS + r)) * HDIM + c2];
    }
    __syncthreads();

    int wid = tid >> 5, lane = tid & 31;
    const float scale = 0.07216878364870322f;   // 1/sqrt(192)

    // 128 score dots, 16 per wave, lane-parallel over H then shfl reduce
    for (int p = wid * 16; p < wid * 16 + 16; ++p) {
        int h = p >> 5, l = p & 31;
        float s = 0.f;
        for (int e = lane; e < HDIM; e += 32)
            s += (float)rows[l * HDIM + e] * wscore[h * HDIM + e];
        s += __shfl_xor(s, 16, 32); s += __shfl_xor(s, 8, 32);
        s += __shfl_xor(s, 4, 32);  s += __shfl_xor(s, 2, 32);
        s += __shfl_xor(s, 1, 32);
        if (lane == 0) sc[p] = (l < len) ? s * scale : -__builtin_inff();
    }
    __syncthreads();

    // masked softmax: one wave per head, lane == l
    if (wid < NHEAD) {
        float v = sc[wid * SPAN_L + lane];
        float mx = v;
        for (int o = 16; o; o >>= 1) mx = fmaxf(mx, __shfl_xor(mx, o, 32));
        float e = (len > 0) ? expf(v - mx) : 0.f;
        float sum = e;
        for (int o = 16; o; o >>= 1) sum += __shfl_xor(sum, o, 32);
        at[wid * SPAN_L + lane] = (sum > 0.f) ? e / sum : 0.f;
    }
    __syncthreads();

    // pooled[h,e] = sum_l attn[h,l] * rows[l,e]
#pragma unroll
    for (int i = 0; i < (NHEAD * HDIM) / 256; ++i) {
        int o = tid + 256 * i;
        int h = o / HDIM, e = o % HDIM;
        float acc = 0.f;
        for (int l = 0; l < len; ++l)
            acc += at[h * SPAN_L + l] * (float)rows[l * HDIM + e];
        pooled[(size_t)bn * (NHEAD * HDIM) + o] = (bf16)acc;
    }
}

// ----------------------------------------------------------- WMMA bf16 GEMM
// C[M,N] = A[M,K] * B[N,K]^T + bias, optional ReLU; f32 and/or bf16 stores.
// REQUIRES: M % 64 == 0, N % 64 == 0, K % 32 == 0 (branch-free inner loop).
// Block: 8 waves; wave = 16(M) x 64(N) tile; block = 64(M) x 128(N).
__global__ __launch_bounds__(256)
void gemm_bf16_wmma(const bf16* __restrict__ A, int lda,
                    const bf16* __restrict__ B, int ldb,
                    const float* __restrict__ bias,
                    float* __restrict__ Cf, int ldcf,
                    bf16* __restrict__ Cb, int ldcb,
                    int M, int N, int K, int relu) {
    int lane = threadIdx.x & 31;
    int wid  = threadIdx.x >> 5;
    int wm = wid & 3, wn = wid >> 2;
    int m0 = blockIdx.y * 64 + wm * 16;
    int n0 = blockIdx.x * 128 + wn * 64;
    if (m0 >= M || n0 >= N) return;   // wave-uniform; surviving waves do all 4 subtiles

    f32x8 acc0, acc1, acc2, acc3;
#pragma unroll
    for (int j = 0; j < 8; ++j) { acc0[j] = 0.f; acc1[j] = 0.f; acc2[j] = 0.f; acc3[j] = 0.f; }

    // A fragment (16-bit A 16x32): lanes 0-15 -> K 0..15, lanes 16-31 -> K 16..31
    int mrow = m0 + (lane & 15);
    int kbA = (lane < 16) ? 0 : 16;
    // B fragment (16-bit B 32x16): lanes 0-15 -> K 0..15, lanes 16-31 -> K 16..31, col = lane&15
    int kbB = (lane < 16) ? 0 : 16;
    int ncol = lane & 15;

    const bf16* Abase  = A + (size_t)mrow * lda + kbA;
    const bf16* Bbase0 = B + (size_t)(n0 +  0 + ncol) * ldb + kbB;
    const bf16* Bbase1 = B + (size_t)(n0 + 16 + ncol) * ldb + kbB;
    const bf16* Bbase2 = B + (size_t)(n0 + 32 + ncol) * ldb + kbB;
    const bf16* Bbase3 = B + (size_t)(n0 + 48 + ncol) * ldb + kbB;

    // software pipeline: preload k=0
    bf16x8 alo = *(const bf16x8*)(Abase);
    bf16x8 ahi = *(const bf16x8*)(Abase + 8);
    bf16x16 b0 = *(const bf16x16*)(Bbase0);
    bf16x16 b1 = *(const bf16x16*)(Bbase1);
    bf16x16 b2 = *(const bf16x16*)(Bbase2);
    bf16x16 b3 = *(const bf16x16*)(Bbase3);

    for (int k = 32; k < K; k += 32) {
        bf16x8 nalo = *(const bf16x8*)(Abase + k);
        bf16x8 nahi = *(const bf16x8*)(Abase + k + 8);
        bf16x16 nb0 = *(const bf16x16*)(Bbase0 + k);
        bf16x16 nb1 = *(const bf16x16*)(Bbase1 + k);
        bf16x16 nb2 = *(const bf16x16*)(Bbase2 + k);
        bf16x16 nb3 = *(const bf16x16*)(Bbase3 + k);

        bf16x16 a;
#pragma unroll
        for (int i = 0; i < 8; ++i) { a[i] = alo[i]; a[8 + i] = ahi[i]; }
        acc0 = __builtin_amdgcn_wmma_f32_16x16x32_bf16(false, a, false, b0, (short)0, acc0, false, false);
        acc1 = __builtin_amdgcn_wmma_f32_16x16x32_bf16(false, a, false, b1, (short)0, acc1, false, false);
        acc2 = __builtin_amdgcn_wmma_f32_16x16x32_bf16(false, a, false, b2, (short)0, acc2, false, false);
        acc3 = __builtin_amdgcn_wmma_f32_16x16x32_bf16(false, a, false, b3, (short)0, acc3, false, false);

        alo = nalo; ahi = nahi; b0 = nb0; b1 = nb1; b2 = nb2; b3 = nb3;
    }
    {   // drain
        bf16x16 a;
#pragma unroll
        for (int i = 0; i < 8; ++i) { a[i] = alo[i]; a[8 + i] = ahi[i]; }
        acc0 = __builtin_amdgcn_wmma_f32_16x16x32_bf16(false, a, false, b0, (short)0, acc0, false, false);
        acc1 = __builtin_amdgcn_wmma_f32_16x16x32_bf16(false, a, false, b1, (short)0, acc1, false, false);
        acc2 = __builtin_amdgcn_wmma_f32_16x16x32_bf16(false, a, false, b2, (short)0, acc2, false, false);
        acc3 = __builtin_amdgcn_wmma_f32_16x16x32_bf16(false, a, false, b3, (short)0, acc3, false, false);
    }

    // C/D layout: VGPR j -> M = m0 + j + (lane<16 ? 0 : 8), N = n0 + nt*16 + (lane&15)
    int moff = (lane & 16) ? 8 : 0;
    f32x8 accs[4] = {acc0, acc1, acc2, acc3};
#pragma unroll
    for (int nt = 0; nt < 4; ++nt) {
        int n = n0 + nt * 16 + ncol;
        float bv = bias ? bias[n] : 0.f;
#pragma unroll
        for (int j = 0; j < 8; ++j) {
            int mm = m0 + j + moff;
            float val = accs[nt][j] + bv;
            if (relu) val = fmaxf(val, 0.f);
            if (Cf) Cf[(size_t)mm * ldcf + n] = val;
            if (Cb) Cb[(size_t)mm * ldcb + n] = (bf16)val;
        }
    }
}

// ---------------------------------------------------------------- LayerNorms
__global__ __launch_bounds__(256)
void ln1_kernel(const float* __restrict__ ao, const float* __restrict__ dq,
                const float* __restrict__ g, const float* __restrict__ bb,
                float* __restrict__ y, bf16* __restrict__ yb) {
    __shared__ float red[256];
    int row = blockIdx.x, tid = threadIdx.x;
    float v[3]; float s = 0.f;
#pragma unroll
    for (int i = 0; i < 3; ++i) {
        int j = tid + 256 * i;
        v[i] = ao[(size_t)row * HDIM + j] + dq[j];
        s += v[i];
    }
    red[tid] = s; __syncthreads();
    for (int o = 128; o; o >>= 1) { if (tid < o) red[tid] += red[tid + o]; __syncthreads(); }
    float mean = red[0] / (float)HDIM; __syncthreads();
    float sq = 0.f;
#pragma unroll
    for (int i = 0; i < 3; ++i) { float d = v[i] - mean; sq += d * d; }
    red[tid] = sq; __syncthreads();
    for (int o = 128; o; o >>= 1) { if (tid < o) red[tid] += red[tid + o]; __syncthreads(); }
    float inv = rsqrtf(red[0] / (float)HDIM + 1e-5f);
#pragma unroll
    for (int i = 0; i < 3; ++i) {
        int j = tid + 256 * i;
        float o = (v[i] - mean) * inv * g[j] + bb[j];
        y[(size_t)row * HDIM + j] = o;
        yb[(size_t)row * HDIM + j] = (bf16)o;
    }
}

__global__ __launch_bounds__(256)
void ln2_kernel(const float* __restrict__ ff, const float* __restrict__ y,
                const unsigned char* __restrict__ masks,
                const float* __restrict__ g, const float* __restrict__ bb,
                float* __restrict__ out) {
    __shared__ float red[256];
    int row = blockIdx.x, tid = threadIdx.x;
    float v[3]; float s = 0.f;
#pragma unroll
    for (int i = 0; i < 3; ++i) {
        int j = tid + 256 * i;
        v[i] = ff[(size_t)row * HDIM + j] + y[(size_t)row * HDIM + j];
        s += v[i];
    }
    red[tid] = s; __syncthreads();
    for (int o = 128; o; o >>= 1) { if (tid < o) red[tid] += red[tid + o]; __syncthreads(); }
    float mean = red[0] / (float)HDIM; __syncthreads();
    float sq = 0.f;
#pragma unroll
    for (int i = 0; i < 3; ++i) { float d = v[i] - mean; sq += d * d; }
    red[tid] = sq; __syncthreads();
    for (int o = 128; o; o >>= 1) { if (tid < o) red[tid] += red[tid + o]; __syncthreads(); }
    float inv = rsqrtf(red[0] / (float)HDIM + 1e-5f);
    float mk = masks[row] ? 1.f : 0.f;
#pragma unroll
    for (int i = 0; i < 3; ++i) {
        int j = tid + 256 * i;
        float o = ((v[i] - mean) * inv * g[j] + bb[j]) * mk;
        out[(size_t)row * HDIM + j] = o;
    }
}

// ------------------------------------------------------------------- launch
extern "C" void kernel_launch(void* const* d_in, const int* in_sizes, int n_in,
                              void* d_out, int out_size, void* d_ws, size_t ws_size,
                              hipStream_t stream) {
    (void)in_sizes; (void)n_in; (void)out_size; (void)ws_size;
    const float* tok   = (const float*)d_in[0];
    const int*   spans = (const int*)d_in[1];
    const unsigned char* masks = (const unsigned char*)d_in[2];
    const float* dq    = (const float*)d_in[4];
    const float* ipw   = (const float*)d_in[5];
    const float* ipb   = (const float*)d_in[6];
    const float* outw  = (const float*)d_in[7];
    const float* outb  = (const float*)d_in[8];
    const float* lng   = (const float*)d_in[9];
    const float* lnb   = (const float*)d_in[10];
    const float* w1    = (const float*)d_in[11];
    const float* b1    = (const float*)d_in[12];
    const float* w2    = (const float*)d_in[13];
    const float* b2    = (const float*)d_in[14];

    char* wp = (char*)d_ws;
    auto alloc = [&](size_t bytes) -> char* {
        char* p = wp; wp += (bytes + 255) & ~(size_t)255; return p;
    };
    float* xf      = (float*)alloc((size_t)ROWS * HDIM * 4);
    bf16*  WvB     = (bf16*) alloc((size_t)HDIM * HDIM * 2);
    bf16*  outWB   = (bf16*) alloc((size_t)HDIM * HDIM * 2);
    bf16*  w1B     = (bf16*) alloc((size_t)FFN_DIM * HDIM * 2);
    bf16*  w2B     = (bf16*) alloc((size_t)HDIM * FFN_DIM * 2);
    float* qv      = (float*)alloc((size_t)HDIM * 4);
    float* wscore  = (float*)alloc((size_t)NHEAD * HDIM * 4);
    bf16*  pooled  = (bf16*) alloc((size_t)ROWS * NHEAD * HDIM * 2); // reused as FFN hidden
    bf16*  ctx     = (bf16*) alloc((size_t)ROWS * HDIM * 2);
    float* attnout = (float*)alloc((size_t)ROWS * HDIM * 4);         // reused as ff
    float* yf      = (float*)alloc((size_t)ROWS * HDIM * 4);
    bf16*  yb      = (bf16*) alloc((size_t)ROWS * HDIM * 2);
    bf16*  hb      = pooled;       // alias: pooled consumed before FFN1 writes
    float* ff      = attnout;      // alias: attn_out consumed by ln1 before FFN2 writes

    // weight converts
    {
        int n = HDIM * HDIM;
        cvt_f32_bf16<<<(n + 255) / 256, 256, 0, stream>>>(ipw + (size_t)2 * HDIM * HDIM, WvB, n);
        cvt_f32_bf16<<<(n + 255) / 256, 256, 0, stream>>>(outw, outWB, n);
        int nf = FFN_DIM * HDIM;
        cvt_f32_bf16<<<(nf + 255) / 256, 256, 0, stream>>>(w1, w1B, nf);
        cvt_f32_bf16<<<(nf + 255) / 256, 256, 0, stream>>>(w2, w2B, nf);
    }
    add_pe_kernel<<<ROWS, 256, 0, stream>>>(tok, xf);
    qproj_kernel<<<3, 256, 0, stream>>>(dq, ipw, ipb, qv);
    wscore_kernel<<<(NHEAD * HDIM + 255) / 256, 256, 0, stream>>>(qv, ipw, wscore);
    attn_pool_kernel<<<ROWS, 256, 0, stream>>>(xf, spans, masks, wscore, pooled);

    // ctx[:, h*192:(h+1)*192] = pooled[:, h, :] @ Wv_h^T + bv_h   (bf16 out); N=192 % 64 == 0
    for (int h = 0; h < NHEAD; ++h) {
        dim3 g((DHEAD + 127) / 128, ROWS / 64);
        gemm_bf16_wmma<<<g, 256, 0, stream>>>(
            pooled + (size_t)h * HDIM, NHEAD * HDIM,
            WvB + (size_t)h * DHEAD * HDIM, HDIM,
            ipb + 2 * HDIM + h * DHEAD,
            nullptr, 0, ctx + (size_t)h * DHEAD, HDIM,
            ROWS, DHEAD, HDIM, 0);
    }
    // attn_out = ctx @ out_w^T + out_b   (f32 out)
    {
        dim3 g(HDIM / 128, ROWS / 64);
        gemm_bf16_wmma<<<g, 256, 0, stream>>>(ctx, HDIM, outWB, HDIM, outb,
                                              attnout, HDIM, nullptr, 0,
                                              ROWS, HDIM, HDIM, 0);
    }
    ln1_kernel<<<ROWS, 256, 0, stream>>>(attnout, dq, lng, lnb, yf, yb);
    // h = relu(y @ w1^T + b1)  (bf16 out)
    {
        dim3 g(FFN_DIM / 128, ROWS / 64);
        gemm_bf16_wmma<<<g, 256, 0, stream>>>(yb, HDIM, w1B, HDIM, b1,
                                              nullptr, 0, hb, FFN_DIM,
                                              ROWS, FFN_DIM, HDIM, 1);
    }
    // ff = h @ w2^T + b2  (f32 out)
    {
        dim3 g(HDIM / 128, ROWS / 64);
        gemm_bf16_wmma<<<g, 256, 0, stream>>>(hb, FFN_DIM, w2B, FFN_DIM, b2,
                                              ff, HDIM, nullptr, 0,
                                              ROWS, HDIM, FFN_DIM, 0);
    }
    ln2_kernel<<<ROWS, 256, 0, stream>>>(ff, yf, masks, lng, lnb, (float*)d_out);
}
